// W8A8O16Linear_59219009077631
// MI455X (gfx1250) — compile-verified, matrix operand order checked by reading, not run on hardware
//
#include <hip/hip_runtime.h>
#include <stdint.h>
#include <stddef.h>

// ---------------------------------------------------------------------------
// W8A8 linear for MI455X (gfx1250, wave32):
//   quantize x,w -> int8 (per-tensor asymmetric), s8*s8 GEMM via
//   v_wmma_i32_16x16x64_iu8 with double-buffered async global->LDS staging,
//   fused dequant + zero-point correction + bias.
// ---------------------------------------------------------------------------

typedef __attribute__((ext_vector_type(8))) int v8i;

#define K_DIM 4096
#define BM 128
#define BN 128
#define BK 64
#define LDST 80   // padded LDS row stride (bytes): 16B-aligned, breaks bank conflicts

// ---------------- CDNA5 async global->LDS copy (ASYNCcnt path) -------------

#if __has_builtin(__builtin_amdgcn_global_load_async_to_lds_b128)
#define ASYNC_COPY 1
#else
#define ASYNC_COPY 0
#endif

// builtin signature (from clang diagnostic): arg0 = AS1 (global) v4i*,
// arg1 = AS3 (LDS) v4i*, arg2 = imm offset, arg3 = imm cpol
typedef int v4i_gcc __attribute__((vector_size(16)));
typedef __attribute__((address_space(1))) v4i_gcc* gptr_v4i;
typedef __attribute__((address_space(3))) v4i_gcc* lptr_v4i;

__device__ __forceinline__ void g2l_b128(void* lds_dst, const void* gsrc) {
#if ASYNC_COPY
    __builtin_amdgcn_global_load_async_to_lds_b128(
        (gptr_v4i)(void*)gsrc, (lptr_v4i)lds_dst, 0, 0);
#else
    *(int4*)lds_dst = *(const int4*)gsrc;
#endif
}

__device__ __forceinline__ void wait_async_copies() {
#if ASYNC_COPY
#if __has_builtin(__builtin_amdgcn_s_wait_asynccnt)
    __builtin_amdgcn_s_wait_asynccnt(0);
#else
    asm volatile("s_wait_asynccnt 0x0" ::: "memory");
#endif
#endif
}

// ---------------- min/max reduction (ordered-uint encoding) ----------------

__device__ __forceinline__ unsigned f2ord(float f) {
    unsigned u = __float_as_uint(f);
    return (u & 0x80000000u) ? ~u : (u | 0x80000000u);
}
__device__ __forceinline__ float ord2f(unsigned e) {
    return __uint_as_float((e & 0x80000000u) ? (e ^ 0x80000000u) : ~e);
}

__global__ void init_slots_kernel(unsigned* __restrict__ slots) {
    // slots: [xmin, xmax, wmin, wmax] (ordered-uint encoded)
    if (threadIdx.x < 4u) slots[threadIdx.x] = (threadIdx.x & 1u) ? 0u : 0xFFFFFFFFu;
}

__global__ __launch_bounds__(256) void minmax_kernel(const float* __restrict__ p, long n,
                                                     unsigned* __restrict__ mn,
                                                     unsigned* __restrict__ mx) {
    __shared__ float smin[256];
    __shared__ float smax[256];
    float lo = 3.4028235e38f, hi = -3.4028235e38f;
    long i = (long)blockIdx.x * blockDim.x + threadIdx.x;
    long stride = (long)gridDim.x * blockDim.x;
    for (; i + 3 * stride < n; i += 4 * stride) {
        float a = p[i], b = p[i + stride], c = p[i + 2 * stride], d = p[i + 3 * stride];
        lo = fminf(lo, fminf(fminf(a, b), fminf(c, d)));
        hi = fmaxf(hi, fmaxf(fmaxf(a, b), fmaxf(c, d)));
    }
    for (; i < n; i += stride) { float v = p[i]; lo = fminf(lo, v); hi = fmaxf(hi, v); }
    smin[threadIdx.x] = lo; smax[threadIdx.x] = hi;
    __syncthreads();
    for (int s = 128; s > 0; s >>= 1) {
        if ((int)threadIdx.x < s) {
            smin[threadIdx.x] = fminf(smin[threadIdx.x], smin[threadIdx.x + s]);
            smax[threadIdx.x] = fmaxf(smax[threadIdx.x], smax[threadIdx.x + s]);
        }
        __syncthreads();
    }
    if (threadIdx.x == 0) {
        atomicMin(mn, f2ord(smin[0]));
        atomicMax(mx, f2ord(smax[0]));
    }
}

// scales layout: [sx, zx, sw, zw, inv_sx, inv_sw]
__global__ void finalize_scales_kernel(const unsigned* __restrict__ slots,
                                       float* __restrict__ sc) {
    float xmin = fminf(ord2f(slots[0]), 0.0f);
    float xmax = fmaxf(ord2f(slots[1]), 0.0f);
    float wmin = fminf(ord2f(slots[2]), 0.0f);
    float wmax = fmaxf(ord2f(slots[3]), 0.0f);
    float sx = fmaxf((xmax - xmin) * (1.0f / 255.0f), 1e-8f);
    float sw = fmaxf((wmax - wmin) * (1.0f / 255.0f), 1e-8f);
    float zx = fminf(fmaxf(rintf(-128.0f - xmin / sx), -128.0f), 127.0f);
    float zw = fminf(fmaxf(rintf(-128.0f - wmin / sw), -128.0f), 127.0f);
    sc[0] = sx; sc[1] = zx; sc[2] = sw; sc[3] = zw;
    sc[4] = 1.0f / sx; sc[5] = 1.0f / sw;
}

// ---------------- quantize rows + per-row sum ----------------

__global__ __launch_bounds__(256) void quantize_rows_kernel(const float* __restrict__ src,
                                                            signed char* __restrict__ q,
                                                            int* __restrict__ rsum,
                                                            const float* __restrict__ sc,
                                                            int which) {
    const float inv_s = sc[4 + which];          // inv_sx or inv_sw
    const float zp    = sc[1 + 2 * which];      // zx or zw
    const long row = blockIdx.x;
    const float* __restrict__ r = src + row * (long)K_DIM;
    signed char* __restrict__ qr = q + row * (long)K_DIM;

    int acc = 0;
    // 4 consecutive elements per thread per step: f32x4 load, packed b32 store
    for (int k4 = (int)threadIdx.x * 4; k4 < K_DIM; k4 += 1024) {
        float4 v = *(const float4*)(r + k4);
        int q0 = (int)fminf(fmaxf(rintf(v.x * inv_s) + zp, -128.0f), 127.0f);
        int q1 = (int)fminf(fmaxf(rintf(v.y * inv_s) + zp, -128.0f), 127.0f);
        int q2 = (int)fminf(fmaxf(rintf(v.z * inv_s) + zp, -128.0f), 127.0f);
        int q3 = (int)fminf(fmaxf(rintf(v.w * inv_s) + zp, -128.0f), 127.0f);
        unsigned packed = (unsigned)(q0 & 0xff) | ((unsigned)(q1 & 0xff) << 8) |
                          ((unsigned)(q2 & 0xff) << 16) | ((unsigned)(q3 & 0xff) << 24);
        *(unsigned*)(qr + k4) = packed;
        acc += q0 + q1 + q2 + q3;
    }
    __shared__ int s[256];
    s[threadIdx.x] = acc;
    __syncthreads();
    for (int st = 128; st > 0; st >>= 1) {
        if ((int)threadIdx.x < st) s[threadIdx.x] += s[threadIdx.x + st];
        __syncthreads();
    }
    if (threadIdx.x == 0) rsum[row] = s[0];
}

// ---------------- IU8 WMMA GEMM ----------------
// Block: 256 threads = 8 waves, tile BM=128 x BN=128, K-step 64, double-buffered
// async global->LDS staging. Waves arranged 4(M) x 2(N): each wave computes a
// 32x64 sub-tile = 2 A-frags x 4 B-frags -> 8 IU8 WMMA per K-step.

union FragU { v8i v; long l[4]; };

__global__ __launch_bounds__(256) void w8a8_gemm_kernel(
    const signed char* __restrict__ qx, const signed char* __restrict__ qw,
    const int* __restrict__ rowsum, const int* __restrict__ colsum,
    const float* __restrict__ bias, const float* __restrict__ sc,
    float* __restrict__ out, int N) {
    __shared__ __align__(16) signed char As[2][BM][LDST];
    __shared__ __align__(16) signed char Bs[2][BN][LDST];

    const int tid  = threadIdx.x;
    const int wave = tid >> 5;
    const int lane = tid & 31;
    const int wm = wave >> 1;         // 0..3 -> M sub-tile
    const int wn = wave & 1;          // 0..1 -> N sub-tile
    const int mBase = blockIdx.y * BM;
    const int nBase = blockIdx.x * BN;

    // cooperative staging: thread -> (row, 16B chunk); 4 b128 copies per thread
    const int ldRow = tid >> 2;          // 0..63
    const int ldVec = (tid & 3) << 4;    // 0,16,32,48
    const signed char* gA0 = qx + (size_t)(mBase + ldRow) * K_DIM + ldVec;
    const signed char* gA1 = gA0 + (size_t)64 * K_DIM;
    const signed char* gB0 = qw + (size_t)(nBase + ldRow) * K_DIM + ldVec;
    const signed char* gB1 = gB0 + (size_t)64 * K_DIM;

    // fragment addressing per ISA 8-bit 16x64 layout:
    // lane<16 holds K bytes {0-7,16-23,32-39,48-55}; lane>=16 offset by +8
    const int fr    = lane & 15;
    const int fhalf = (lane >> 4) << 3;  // 0 or 8

    v8i acc[2][4] = {};

    // prologue: stage K-tile 0 into buffer 0
    g2l_b128(&As[0][ldRow][ldVec],      gA0);
    g2l_b128(&As[0][ldRow + 64][ldVec], gA1);
    g2l_b128(&Bs[0][ldRow][ldVec],      gB0);
    g2l_b128(&Bs[0][ldRow + 64][ldVec], gB1);

    for (int kb = 0; kb < K_DIM; kb += BK) {
        const int cur = (kb >> 6) & 1;
        wait_async_copies();   // my copies into buf[cur] are done
        __syncthreads();       // everyone's copies done; everyone done reading buf[cur^1]

        if (kb + BK < K_DIM) {  // stage next tile into the other buffer (overlaps compute)
            const int nxt = cur ^ 1;
            const int kn = kb + BK;
            g2l_b128(&As[nxt][ldRow][ldVec],      gA0 + kn);
            g2l_b128(&As[nxt][ldRow + 64][ldVec], gA1 + kn);
            g2l_b128(&Bs[nxt][ldRow][ldVec],      gB0 + kn);
            g2l_b128(&Bs[nxt][ldRow + 64][ldVec], gB1 + kn);
        }

        // A fragments: rows wm*32 + {0,16}
        const signed char* ap = &As[cur][wm * 32 + fr][fhalf];
        FragU a0, a1;
        a0.l[0] = *(const long*)(ap + 0);
        a0.l[1] = *(const long*)(ap + 16);
        a0.l[2] = *(const long*)(ap + 32);
        a0.l[3] = *(const long*)(ap + 48);
        ap += 16 * LDST;
        a1.l[0] = *(const long*)(ap + 0);
        a1.l[1] = *(const long*)(ap + 16);
        a1.l[2] = *(const long*)(ap + 32);
        a1.l[3] = *(const long*)(ap + 48);

#pragma unroll
        for (int j = 0; j < 4; ++j) {
            const signed char* bp = &Bs[cur][wn * 64 + j * 16 + fr][fhalf];
            FragU b;
            b.l[0] = *(const long*)(bp + 0);
            b.l[1] = *(const long*)(bp + 16);
            b.l[2] = *(const long*)(bp + 32);
            b.l[3] = *(const long*)(bp + 48);
            // signed x signed int8, i32 accumulate
            acc[0][j] = __builtin_amdgcn_wmma_i32_16x16x64_iu8(
                true, a0.v, true, b.v, acc[0][j], false, false);
            acc[1][j] = __builtin_amdgcn_wmma_i32_16x16x64_iu8(
                true, a1.v, true, b.v, acc[1][j], false, false);
        }
    }

    // epilogue: dequant with zero-point correction + bias
    const float sx = sc[0], zx = sc[1], sw = sc[2], zw = sc[3];
    const float ss  = sx * sw;
    const float kzz = (float)K_DIM * zx * zw;

#pragma unroll
    for (int a = 0; a < 2; ++a) {
        const int mRow = mBase + wm * 32 + a * 16 + ((lane >> 4) << 3);  // + v gives M
#pragma unroll
        for (int j = 0; j < 4; ++j) {
            const int n = nBase + wn * 64 + j * 16 + (lane & 15);
            const float csn = (float)colsum[n];
            const float bi  = bias[n];
#pragma unroll
            for (int v = 0; v < 8; ++v) {
                const int m = mRow + v;
                const float a32 = (float)acc[a][j][v];
                const float val = ss * (a32 - zx * csn - zw * (float)rowsum[m] + kzz) + bi;
                out[(size_t)m * (size_t)N + (size_t)n] = val;
            }
        }
    }
}

// ---------------- host launcher ----------------

extern "C" void kernel_launch(void* const* d_in, const int* in_sizes, int n_in,
                              void* d_out, int out_size, void* d_ws, size_t ws_size,
                              hipStream_t stream) {
    const float* x    = (const float*)d_in[0];   // [M, K]
    const float* w    = (const float*)d_in[1];   // [N, K]
    const float* bias = (const float*)d_in[2];   // [N]
    float* out = (float*)d_out;

    const long K = K_DIM;
    const long M = (long)in_sizes[0] / K;   // 8192
    const long N = (long)in_sizes[2];       // 11008

    // workspace carve-up
    char* ws = (char*)d_ws;
    unsigned* slots  = (unsigned*)(ws + 0);   // 4 encoded min/max
    float*    scales = (float*)(ws + 16);     // sx,zx,sw,zw,inv_sx,inv_sw
    int*      rowsum = (int*)(ws + 64);
    size_t offCol = 64 + (size_t)M * 4;
    int*      colsum = (int*)(ws + offCol);
    size_t offQX = (offCol + (size_t)N * 4 + 255) & ~(size_t)255;
    signed char* qx = (signed char*)(ws + offQX);
    size_t offQW = (offQX + (size_t)(M * K) + 255) & ~(size_t)255;
    signed char* qw = (signed char*)(ws + offQW);

    // 1) reset atomic slots every call (graph replay safe)
    init_slots_kernel<<<1, 32, 0, stream>>>(slots);

    // 2) global min/max of x and w
    minmax_kernel<<<2048, 256, 0, stream>>>(x, M * K, slots + 0, slots + 1);
    minmax_kernel<<<2048, 256, 0, stream>>>(w, N * K, slots + 2, slots + 3);

    // 3) scale / zero-point
    finalize_scales_kernel<<<1, 1, 0, stream>>>(slots, scales);

    // 4) quantize + per-row sums
    quantize_rows_kernel<<<(unsigned)M, 256, 0, stream>>>(x, qx, rowsum, scales, 0);
    quantize_rows_kernel<<<(unsigned)N, 256, 0, stream>>>(w, qw, colsum, scales, 1);

    // 5) int8 GEMM + fused dequant epilogue (N = 86*128, M = 64*128 -> exact tiling)
    dim3 grid((unsigned)(N / BN), (unsigned)(M / BM));
    w8a8_gemm_kernel<<<grid, 256, 0, stream>>>(qx, qw, rowsum, colsum, bias, scales,
                                               out, (int)N);
}
